// GATv2_85822036509193
// MI455X (gfx1250) — matrix operand order repaired and checked by reading, not compiled, not run on hardware
//
#include <hip/hip_runtime.h>
#include <hip/hip_bf16.h>
#include <math.h>

typedef __attribute__((ext_vector_type(16))) __bf16 v16bf;
typedef __attribute__((ext_vector_type(8)))  float  v8f;

#define TPB 256
#define NEG_SLOPE 0.2f

// GEMM blocking: 8 waves/block, each wave owns a 16x64 output strip.
#define BM 128            // rows per block (8 waves * 16)
#define BN 64             // cols per block (4 WMMA tiles per wave)
#define KMAX 256          // max K across layers

static __device__ __forceinline__ unsigned f2ord(float f) {
  unsigned u = __float_as_uint(f);
  return (u & 0x80000000u) ? ~u : (u | 0x80000000u);
}
static __device__ __forceinline__ float ord2f(unsigned u) {
  return __uint_as_float((u & 0x80000000u) ? (u & 0x7fffffffu) : ~u);
}
static __device__ __forceinline__ float leaky(float v) {
  return v > 0.0f ? v : NEG_SLOPE * v;
}

// ---------------- utility kernels ----------------
__global__ void fill_f32(float* p, float v, int n) {
  int i = blockIdx.x * blockDim.x + threadIdx.x;
  if (i < n) p[i] = v;
}
__global__ void fill_u32(unsigned* p, unsigned v, int n) {
  int i = blockIdx.x * blockDim.x + threadIdx.x;
  if (i < n) p[i] = v;
}
__global__ void cvt_f32_bf16(const float* __restrict__ s, __bf16* __restrict__ d, int n) {
  int i = blockIdx.x * blockDim.x + threadIdx.x;
  if (i < n) d[i] = (__bf16)s[i];
}
// Wt[c][k] = W[k][c]  (bf16, transposed so B fragment words are K-contiguous)
__global__ void cvt_w_transpose(const float* __restrict__ W, __bf16* __restrict__ Wt,
                                int K, int Nout) {
  int i = blockIdx.x * blockDim.x + threadIdx.x;
  if (i < K * Nout) {
    int k = i / Nout, c = i - k * Nout;
    Wt[(size_t)c * K + k] = (__bf16)W[i];
  }
}

// ---------------- dense phase: LDS-staged bf16 WMMA GEMM ----------------
// C[M,Nout] = A[M,K] x B[K,Nout]; A row-major bf16; B pre-transposed (Bt[Nout,K]).
// Block: 256 thr = 8 waves; B tile (BN x K) staged in LDS once, then each wave
// runs 4 independent v_wmma_f32_16x16x32_bf16 per K-step off one A fragment.
__global__ void gemm_bf16_wmma(const __bf16* __restrict__ A,
                               const __bf16* __restrict__ Bt,
                               float* __restrict__ C,
                               int M, int K, int Nout, int groupsN) {
  __shared__ __bf16 ldsB[BN * KMAX];      // 32 KB max
  int bm = blockIdx.x / groupsN;
  int bn = blockIdx.x - bm * groupsN;
  int colBase = bn * BN;

  // cooperative contiguous copy of the B tile into LDS (b128 in, b128 out)
  {
    const uint4* src = (const uint4*)(Bt + (size_t)colBase * K);
    uint4*       dst = (uint4*)ldsB;
    int nvec = (BN * K) >> 3;             // 16B vectors
    for (int i = threadIdx.x; i < nvec; i += blockDim.x) dst[i] = src[i];
  }
  __syncthreads();

  int wv   = threadIdx.x >> 5;
  int lane = threadIdx.x & 31;
  int half = lane >> 4;                   // K sub-range select per lane half
  int l    = lane & 15;
  int tm   = bm * (BM / 16) + wv;
  if (tm * 16 >= M) return;               // wave-uniform: EXEC stays all-ones
  int row = tm * 16 + l;
  const __bf16* arow = A + (size_t)row * K;

  union AB { uint4 q[2]; v16bf v; };
  v8f acc[4] = {};

  for (int kb = 0; kb < K; kb += 32) {
    // A 16x32 bf16 fragment: lane half h holds K kb+8h..+7 and kb+16+8h..+7
    AB a;
    const uint4* ap = (const uint4*)(arow + kb + half * 8);
    a.q[0] = ap[0];                       // K = kb+8h   .. +7
    a.q[1] = ap[2];                       // K = kb+16+8h.. +7  (+32 bytes)
    __builtin_prefetch(arow + kb + 64, 0, 3);
#pragma unroll
    for (int t = 0; t < 4; ++t) {
      // B 32x16 fragment: for column colLocal the 8 words are K-contiguous
      AB b;
      const uint4* bp = (const uint4*)(ldsB + (size_t)(t * 16 + l) * K + kb + half * 8);
      b.q[0] = bp[0];
      b.q[1] = bp[2];
      acc[t] = __builtin_amdgcn_wmma_f32_16x16x32_bf16(false, a.v, false, b.v,
                                                       (short)0, acc[t], false, false);
    }
  }

  // D layout: VGPR r -> M = half*8 + r, N = lane&15
  int baseRow = tm * 16 + half * 8;
#pragma unroll
  for (int t = 0; t < 4; ++t) {
#pragma unroll
    for (int r = 0; r < 8; ++r) {
      int rr = baseRow + r;
      if (rr < M) C[(size_t)rr * Nout + colBase + t * 16 + l] = acc[t][r];
    }
  }
}

// ---------------- attention scalars ----------------
__global__ void attn_logits(const float* __restrict__ Hb,
                            const float* __restrict__ asrc, const float* __restrict__ adst,
                            float* __restrict__ alsrc, float* __restrict__ aldst,
                            int N, int Hh, int C) {
  int tid = blockIdx.x * blockDim.x + threadIdx.x;
  if (tid >= N * Hh) return;
  int n = tid / Hh, h = tid - n * Hh;
  const float* hp = Hb + (size_t)n * Hh * C + (size_t)h * C;
  const float* as = asrc + h * C;
  const float* ad = adst + h * C;
  float s0 = 0.f, s1 = 0.f;
  for (int c = 0; c < C; ++c) { float v = hp[c]; s0 += v * as[c]; s1 += v * ad[c]; }
  alsrc[tid] = s0; aldst[tid] = s1;
}

// ---------------- sparse phase: 3 edge passes (edges + self loops) ----------------
__global__ void edge_max_kernel(const int* __restrict__ src, const int* __restrict__ dst,
                                const float* __restrict__ alsrc, const float* __restrict__ aldst,
                                unsigned* __restrict__ mbits, int E, int N, int Hh) {
  int tid = blockIdx.x * blockDim.x + threadIdx.x;
  int Etot = E + N;
  if (tid >= Etot * Hh) return;
  int i = tid / Hh, h = tid - i * Hh;
  int s, d;
  if (i < E) { s = src[i]; d = dst[i]; } else { s = d = i - E; }
  float e = leaky(alsrc[s * Hh + h] + aldst[d * Hh + h]);
  atomicMax(&mbits[d * Hh + h], f2ord(e));
}

__global__ void edge_expsum_kernel(const int* __restrict__ src, const int* __restrict__ dst,
                                   const float* __restrict__ alsrc, const float* __restrict__ aldst,
                                   const unsigned* __restrict__ mbits,
                                   float* __restrict__ denom, int E, int N, int Hh) {
  int tid = blockIdx.x * blockDim.x + threadIdx.x;
  int Etot = E + N;
  if (tid >= Etot * Hh) return;
  int i = tid / Hh, h = tid - i * Hh;
  int s, d;
  if (i < E) { s = src[i]; d = dst[i]; } else { s = d = i - E; }
  float e  = leaky(alsrc[s * Hh + h] + aldst[d * Hh + h]);
  float mm = ord2f(mbits[d * Hh + h]);
  atomicAdd(&denom[d * Hh + h], expf(e - mm));
}

__global__ void edge_aggregate_kernel(const int* __restrict__ src, const int* __restrict__ dst,
                                      const float* __restrict__ alsrc, const float* __restrict__ aldst,
                                      const unsigned* __restrict__ mbits,
                                      const float* __restrict__ denom,
                                      const float* __restrict__ Hb,
                                      float* __restrict__ agg,
                                      int E, int N, int Hh, int C) {
  int tid = blockIdx.x * blockDim.x + threadIdx.x;
  int Etot = E + N;
  if (tid >= Etot * Hh) return;
  int i = tid / Hh, h = tid - i * Hh;
  int s, d;
  if (i < E) { s = src[i]; d = dst[i]; } else { s = d = i - E; }
  float e  = leaky(alsrc[s * Hh + h] + aldst[d * Hh + h]);
  float mm = ord2f(mbits[d * Hh + h]);
  float alpha = expf(e - mm) / (denom[d * Hh + h] + 1e-16f);
  const float* hs = Hb  + (size_t)s * Hh * C + (size_t)h * C;
  float*       ag = agg + (size_t)d * Hh * C + (size_t)h * C;
  for (int c = 0; c < C; ++c) atomicAdd(&ag[c], hs[c] * alpha);
}

__global__ void bias_elu_inplace(float* __restrict__ a, const float* __restrict__ b,
                                 int N, int HC) {
  int i = blockIdx.x * blockDim.x + threadIdx.x;
  if (i >= N * HC) return;
  float v = a[i] + b[i % HC];
  a[i] = v > 0.f ? v : (expf(v) - 1.f);
}

// ---------------- pooling + head ----------------
__global__ void pool_kernel(const float* __restrict__ h, const int* __restrict__ batch,
                            float* __restrict__ sums, float* __restrict__ cnts, int N) {
  int i = blockIdx.x * blockDim.x + threadIdx.x;
  if (i >= N * 64) return;
  int n = i >> 6, c = i & 63;
  int g = batch[n];
  atomicAdd(&sums[g * 64 + c], h[i]);
  if (c == 0) atomicAdd(&cnts[g], 1.0f);
}

__global__ void head_kernel(const float* __restrict__ sums, const float* __restrict__ cnts,
                            const float* __restrict__ linW, const float* __restrict__ linb,
                            float* __restrict__ out, int G, int OUT, int C) {
  int i = blockIdx.x * blockDim.x + threadIdx.x;
  if (i >= G * OUT) return;
  int g = i / OUT, o = i - g * OUT;
  float cnt = fmaxf(cnts[g], 1.0f);
  float acc = linb[o];
  for (int c = 0; c < C; ++c) acc += (sums[g * C + c] / cnt) * linW[c * OUT + o];
  out[i] = acc;
}

// ---------------- host-side helpers ----------------
static inline int cdiv(int a, int b) { return (a + b - 1) / b; }

static void gat_layer(const float* Fin, int N, int K,
                      const float* W, const float* bias,
                      const float* asrc, const float* adst,
                      int Hh, int C,
                      const int* srcA, const int* dstA, int E,
                      __bf16* Xh, __bf16* Wt, float* Hb, float* agg,
                      float* alsrc, float* aldst, unsigned* mbits, float* denom,
                      hipStream_t stream) {
  int HC = Hh * C;
  // stage bf16 operands
  cvt_f32_bf16<<<cdiv(N * K, TPB), TPB, 0, stream>>>(Fin, Xh, N * K);
  cvt_w_transpose<<<cdiv(K * HC, TPB), TPB, 0, stream>>>(W, Wt, K, HC);
  // dense transform on WMMA
  int groupsM = cdiv(N, BM);
  int groupsN = HC / BN;                  // 4 for 256-wide layers, 1 for layer 3
  gemm_bf16_wmma<<<groupsM * groupsN, TPB, 0, stream>>>(Xh, Wt, Hb, N, K, HC, groupsN);
  // attention scalars
  attn_logits<<<cdiv(N * Hh, TPB), TPB, 0, stream>>>(Hb, asrc, adst, alsrc, aldst, N, Hh, C);
  // reset reductions (graph-replay safe: re-init every call)
  fill_u32<<<cdiv(N * Hh, TPB), TPB, 0, stream>>>(mbits, 0u, N * Hh);
  fill_f32<<<cdiv(N * Hh, TPB), TPB, 0, stream>>>(denom, 0.f, N * Hh);
  fill_f32<<<cdiv(N * HC, TPB), TPB, 0, stream>>>(agg, 0.f, N * HC);
  // segment softmax + aggregate over edges (incl. self loops)
  int Etot = E + N;
  edge_max_kernel<<<cdiv(Etot * Hh, TPB), TPB, 0, stream>>>(srcA, dstA, alsrc, aldst,
                                                            mbits, E, N, Hh);
  edge_expsum_kernel<<<cdiv(Etot * Hh, TPB), TPB, 0, stream>>>(srcA, dstA, alsrc, aldst,
                                                               mbits, denom, E, N, Hh);
  edge_aggregate_kernel<<<cdiv(Etot * Hh, TPB), TPB, 0, stream>>>(srcA, dstA, alsrc, aldst,
                                                                  mbits, denom, Hb, agg,
                                                                  E, N, Hh, C);
  bias_elu_inplace<<<cdiv(N * HC, TPB), TPB, 0, stream>>>(agg, bias, N, HC);
}

extern "C" void kernel_launch(void* const* d_in, const int* in_sizes, int n_in,
                              void* d_out, int out_size, void* d_ws, size_t ws_size,
                              hipStream_t stream) {
  // inputs per setup_inputs() order
  const float* x     = (const float*)d_in[0];
  const int*   ei    = (const int*)  d_in[1];
  const int*   batch = (const int*)  d_in[2];
  const float* W1    = (const float*)d_in[3];
  const float* b1    = (const float*)d_in[4];
  const float* as1   = (const float*)d_in[5];
  const float* ad1   = (const float*)d_in[6];
  const float* W2    = (const float*)d_in[7];
  const float* b2    = (const float*)d_in[8];
  const float* as2   = (const float*)d_in[9];
  const float* ad2   = (const float*)d_in[10];
  const float* W3    = (const float*)d_in[11];
  const float* b3    = (const float*)d_in[12];
  const float* as3   = (const float*)d_in[13];
  const float* ad3   = (const float*)d_in[14];
  const float* linW  = (const float*)d_in[15];
  const float* linb  = (const float*)d_in[16];

  const int N  = in_sizes[2];          // 50000
  const int E  = in_sizes[1] / 2;      // 600000
  const int IN = in_sizes[0] / N;      // 128
  const int HID = 64, HEADS = 4, G = 64, OUT = 10;
  const int HC = HEADS * HID;          // 256

  const int* srcA = ei;
  const int* dstA = ei + E;

  // carve workspace
  char* w = (char*)d_ws;
  size_t off = 0;
  auto carve = [&](size_t bytes) -> void* {
    void* p = w + off;
    off = (off + bytes + 255) & ~(size_t)255;
    return p;
  };
  __bf16*   Xh    = (__bf16*)  carve((size_t)N * HC * 2);      // bf16 activations (K<=256)
  __bf16*   Wt    = (__bf16*)  carve((size_t)HC * HC * 2);     // bf16 transposed weights
  float*    B0    = (float*)   carve((size_t)N * HC * 4);      // ping-pong node buffer
  float*    B1    = (float*)   carve((size_t)N * HC * 4);      // ping-pong node buffer
  float*    Hb    = (float*)   carve((size_t)N * HC * 4);      // GEMM output h
  float*    alsrc = (float*)   carve((size_t)N * HEADS * 4);
  float*    aldst = (float*)   carve((size_t)N * HEADS * 4);
  unsigned* mbits = (unsigned*)carve((size_t)N * HEADS * 4);
  float*    denom = (float*)   carve((size_t)N * HEADS * 4);
  float*    sums  = (float*)   carve((size_t)G * HID * 4);
  float*    cnts  = (float*)   carve((size_t)G * 4);

  // layer 1: x[ N,128 ] -> B0[ N,256 ]
  gat_layer(x,  N, IN, W1, b1, as1, ad1, HEADS, HID, srcA, dstA, E,
            Xh, Wt, Hb, B0, alsrc, aldst, mbits, denom, stream);
  // layer 2: B0[ N,256 ] -> B1[ N,256 ]
  gat_layer(B0, N, HC, W2, b2, as2, ad2, HEADS, HID, srcA, dstA, E,
            Xh, Wt, Hb, B1, alsrc, aldst, mbits, denom, stream);
  // layer 3: B1[ N,256 ] -> B0[ N,64 ]  (single head)
  gat_layer(B1, N, HC, W3, b3, as3, ad3, 1, HID, srcA, dstA, E,
            Xh, Wt, Hb, B0, alsrc, aldst, mbits, denom, stream);

  // global mean pool + linear head
  fill_f32<<<cdiv(G * HID, TPB), TPB, 0, stream>>>(sums, 0.f, G * HID);
  fill_f32<<<1, TPB, 0, stream>>>(cnts, 0.f, G);
  pool_kernel<<<cdiv(N * HID, TPB), TPB, 0, stream>>>(B0, batch, sums, cnts, N);
  head_kernel<<<cdiv(G * OUT, TPB), TPB, 0, stream>>>(sums, cnts, linW, linb,
                                                      (float*)d_out, G, OUT, HID);
}
